// MaskedAttention_40750649704655
// MI455X (gfx1250) — matrix operand order repaired and checked by reading, not compiled, not run on hardware
//
#include <hip/hip_runtime.h>
#include <hip/hip_bf16.h>

// ---------------- types / constants ----------------
typedef __attribute__((ext_vector_type(16))) __bf16 v16bf;
typedef __attribute__((ext_vector_type(8)))  float  v8f;

#define B_  4
#define T_  2048
#define D_  1024
#define MT_ (B_ * T_)           // 8192 flattened rows
#define SCALE_ 0.03125f         // 1/sqrt(1024)
#define NEG_BIG (-3.0e38f)

union FragBF { v16bf v; unsigned int d[8]; };
union AccU   { v8f v; float f[8]; };

__device__ inline unsigned short f2bf(float f) {
    unsigned int u = __float_as_uint(f);
    u += 0x7FFFu + ((u >> 16) & 1u);      // round-to-nearest-even
    return (unsigned short)(u >> 16);
}

// A fragment: 16x32 bf16 tile, row-major source (row=M, k contiguous).
// ISA layout: lanes 0-15 -> M=lane, K groups {0..7,16..23}; lanes 16-31 -> +8 in K.
__device__ inline void load_fragA(FragBF& fa, const unsigned short* A, int lda, int lane) {
    const unsigned short* p = A + (lane & 15) * lda + ((lane >> 4) << 3);
#pragma unroll
    for (int r = 0; r < 8; ++r) {
        int k = ((r & 4) << 2) + ((r & 3) << 1);   // (r<4?0:16) + 2*(r&3)
        fa.d[r] = *(const unsigned int*)(p + k);
    }
}

// B fragment: 32x16 bf16 tile from an N-major buffer Bt (row = output col n, k contiguous).
// ISA layout: lanes 0-15 -> N=lane, K groups {0..7,16..23}; lanes 16-31 -> +8 in K.
__device__ inline void load_fragB(FragBF& fb, const unsigned short* Bt, int ldb, int lane) {
    const unsigned short* p = Bt + (lane & 15) * ldb + ((lane >> 4) << 3);
#pragma unroll
    for (int r = 0; r < 8; ++r)
        fb.d[r] = *(const unsigned int*)(p + ((r & 4) << 2) + ((r & 3) << 1));
}

__device__ inline v8f wmma_bf(const FragBF& a, const FragBF& b, v8f c) {
    return __builtin_amdgcn_wmma_f32_16x16x32_bf16(false, a.v, false, b.v,
                                                   (short)0, c, false, false);
}

// wave computes a 32(M) x 64(N) tile: 2 A-frags x 4 B-frags, K-loop step 32
__device__ inline void wave_gemm(const unsigned short* A, int lda,
                                 const unsigned short* Bt, int ldb,
                                 int Kd, int lane, AccU acc[2][4]) {
    for (int k0 = 0; k0 < Kd; k0 += 32) {
        FragBF fa0, fa1;
        load_fragA(fa0, A + k0, lda, lane);
        load_fragA(fa1, A + 16 * lda + k0, lda, lane);
#pragma unroll
        for (int ni = 0; ni < 4; ++ni) {
            FragBF fb;
            load_fragB(fb, Bt + (ni * 16) * ldb + k0, ldb, lane);
            acc[0][ni].v = wmma_bf(fa0, fb, acc[0][ni].v);
            acc[1][ni].v = wmma_bf(fa1, fb, acc[1][ni].v);
        }
    }
}

// ---------------- conversion kernels ----------------
__global__ __launch_bounds__(256) void k_cvt_bf16(const float* __restrict__ in,
                                                  unsigned short* __restrict__ out, int n) {
    int i = blockIdx.x * 256 + threadIdx.x;
    if (i < n) out[i] = f2bf(in[i]);
}

// W[k][n] f32 -> Wt[n][k] bf16 (N-major for K-contiguous B fragments)
__global__ __launch_bounds__(256) void k_transpose_bf16(const float* __restrict__ W,
                                                        unsigned short* __restrict__ Wt) {
    int n = blockIdx.x * 16 + threadIdx.x;
    int k = blockIdx.y * 16 + threadIdx.y;
    Wt[n * D_ + k] = f2bf(W[k * D_ + n]);
}

// ---------------- projection GEMM: C = x16 @ Wt^T + bias -> bf16 ----------------
// OutT != nullptr  => store transposed per batch (V path: Vt[b][d][t])
__global__ __launch_bounds__(128) void k_proj(const unsigned short* __restrict__ A,
                                              const unsigned short* __restrict__ Wt,
                                              const float* __restrict__ bias,
                                              unsigned short* __restrict__ Out,
                                              unsigned short* __restrict__ OutT) {
    const int lane = threadIdx.x & 31;
    const int wv   = threadIdx.x >> 5;
    const int row0 = blockIdx.y * 64 + (wv >> 1) * 32;
    const int col0 = blockIdx.x * 128 + (wv & 1) * 64;

    AccU acc[2][4];
#pragma unroll
    for (int mi = 0; mi < 2; ++mi)
#pragma unroll
        for (int ni = 0; ni < 4; ++ni)
#pragma unroll
            for (int i = 0; i < 8; ++i) acc[mi][ni].f[i] = 0.f;

    wave_gemm(A + (size_t)row0 * D_, D_, Wt + (size_t)col0 * D_, D_, D_, lane, acc);

    const int half = lane >> 4, cl = lane & 15;
#pragma unroll
    for (int mi = 0; mi < 2; ++mi)
#pragma unroll
        for (int ni = 0; ni < 4; ++ni) {
            int col = col0 + ni * 16 + cl;
            float bb = bias[col];
#pragma unroll
            for (int r = 0; r < 8; ++r) {
                int row = row0 + mi * 16 + r + 8 * half;
                float v = acc[mi][ni].f[r] + bb;
                if (OutT) {
                    int b = row >> 11, t = row & (T_ - 1);
                    OutT[((size_t)b * D_ + col) * T_ + t] = f2bf(v);
                } else {
                    Out[(size_t)row * D_ + col] = f2bf(v);
                }
            }
        }
}

// ---------------- pass 1: causal softmax row stats (online max/sumexp) ----------------
__global__ __launch_bounds__(128) void k_stats(const unsigned short* __restrict__ Q,
                                               const unsigned short* __restrict__ K,
                                               float* __restrict__ m_out,
                                               float* __restrict__ l_out) {
    const int lane = threadIdx.x & 31;
    const int wv   = threadIdx.x >> 5;
    const int b    = blockIdx.y;
    const int t0   = (blockIdx.x * 4 + wv) * 16;
    const int half = lane >> 4, cl = lane & 15;

    const unsigned short* Qb = Q + ((size_t)b * T_ + t0) * D_;
    const unsigned short* Kb = K + (size_t)b * T_ * D_;

    float m_run[8], l_run[8];
#pragma unroll
    for (int r = 0; r < 8; ++r) { m_run[r] = NEG_BIG; l_run[r] = 0.f; }

    for (int s0 = 0; s0 <= t0 + 15; s0 += 16) {
        AccU acc;
#pragma unroll
        for (int i = 0; i < 8; ++i) acc.f[i] = 0.f;
        for (int k0 = 0; k0 < D_; k0 += 32) {
            FragBF fa, fb;
            load_fragA(fa, Qb + k0, D_, lane);
            load_fragB(fb, Kb + (size_t)s0 * D_ + k0, D_, lane);
            acc.v = wmma_bf(fa, fb, acc.v);
        }
#pragma unroll
        for (int r = 0; r < 8; ++r) {
            int row = t0 + r + 8 * half;
            int col = s0 + cl;
            float v = acc.f[r] * SCALE_;
            if (col > row) v = NEG_BIG;
            float tmax = v;
            tmax = fmaxf(tmax, __shfl_xor(tmax, 1));
            tmax = fmaxf(tmax, __shfl_xor(tmax, 2));
            tmax = fmaxf(tmax, __shfl_xor(tmax, 4));
            tmax = fmaxf(tmax, __shfl_xor(tmax, 8));
            float mnew = fmaxf(m_run[r], tmax);
            float p = __expf(v - mnew);
            float ps = p;
            ps += __shfl_xor(ps, 1);
            ps += __shfl_xor(ps, 2);
            ps += __shfl_xor(ps, 4);
            ps += __shfl_xor(ps, 8);
            l_run[r] = l_run[r] * __expf(m_run[r] - mnew) + ps;
            m_run[r] = mnew;
        }
    }
#pragma unroll
    for (int r = 0; r < 8; ++r) {
        if (cl == r) {
            int row = t0 + r + 8 * half;
            m_out[b * T_ + row] = m_run[r];
            l_out[b * T_ + row] = l_run[r];
        }
    }
}

// ---------------- pass 2: materialize P = softmax(S) in bf16 ----------------
__global__ __launch_bounds__(128) void k_probs(const unsigned short* __restrict__ Q,
                                               const unsigned short* __restrict__ K,
                                               const float* __restrict__ m_in,
                                               const float* __restrict__ l_in,
                                               unsigned short* __restrict__ P) {
    const int lane  = threadIdx.x & 31;
    const int wv    = threadIdx.x >> 5;
    const int b     = blockIdx.z;
    const int ttile = blockIdx.x;
    const int stile = blockIdx.y * 4 + wv;
    const int t0 = ttile * 16, s0 = stile * 16;
    const int half = lane >> 4, cl = lane & 15;

    if (stile > ttile) {  // fully masked tile: zero-fill
#pragma unroll
        for (int r = 0; r < 8; ++r) {
            int row = t0 + r + 8 * half;
            P[((size_t)b * T_ + row) * T_ + s0 + cl] = 0;
        }
        return;
    }

    const unsigned short* Qb = Q + ((size_t)b * T_ + t0) * D_;
    const unsigned short* Kb = K + ((size_t)b * T_ + s0) * D_;
    AccU acc;
#pragma unroll
    for (int i = 0; i < 8; ++i) acc.f[i] = 0.f;
    for (int k0 = 0; k0 < D_; k0 += 32) {
        FragBF fa, fb;
        load_fragA(fa, Qb + k0, D_, lane);
        load_fragB(fb, Kb + k0, D_, lane);
        acc.v = wmma_bf(fa, fb, acc.v);
    }
#pragma unroll
    for (int r = 0; r < 8; ++r) {
        int row = t0 + r + 8 * half;
        int col = s0 + cl;
        float mm = m_in[b * T_ + row];
        float ll = l_in[b * T_ + row];
        float v = acc.f[r] * SCALE_;
        float p = (col <= row) ? (__expf(v - mm) / ll) : 0.f;
        P[((size_t)b * T_ + row) * T_ + col] = f2bf(p);
    }
}

// ---------------- attn @ V  (per batch, Vt is [D][T]) -> bf16 ----------------
__global__ __launch_bounds__(128) void k_pv(const unsigned short* __restrict__ P,
                                            const unsigned short* __restrict__ Vt,
                                            unsigned short* __restrict__ AttnOut) {
    const int lane = threadIdx.x & 31;
    const int wv   = threadIdx.x >> 5;
    const int b    = blockIdx.z;
    const int row0 = blockIdx.y * 64 + (wv >> 1) * 32;
    const int col0 = blockIdx.x * 128 + (wv & 1) * 64;

    AccU acc[2][4];
#pragma unroll
    for (int mi = 0; mi < 2; ++mi)
#pragma unroll
        for (int ni = 0; ni < 4; ++ni)
#pragma unroll
            for (int i = 0; i < 8; ++i) acc[mi][ni].f[i] = 0.f;

    wave_gemm(P + (size_t)b * T_ * T_ + (size_t)row0 * T_, T_,
              Vt + (size_t)b * D_ * T_ + (size_t)col0 * T_, T_,
              T_, lane, acc);

    const int half = lane >> 4, cl = lane & 15;
#pragma unroll
    for (int mi = 0; mi < 2; ++mi)
#pragma unroll
        for (int ni = 0; ni < 4; ++ni) {
            int col = col0 + ni * 16 + cl;
#pragma unroll
            for (int r = 0; r < 8; ++r) {
                int row = row0 + mi * 16 + r + 8 * half;
                AttnOut[((size_t)b * T_ + row) * D_ + col] = f2bf(acc[mi][ni].f[r]);
            }
        }
}

// ---------------- final projection: out = attn @ Wh + bh  (f32 out) ----------------
__global__ __launch_bounds__(128) void k_final(const unsigned short* __restrict__ A,
                                               const unsigned short* __restrict__ Wt,
                                               const float* __restrict__ bias,
                                               float* __restrict__ Out) {
    const int lane = threadIdx.x & 31;
    const int wv   = threadIdx.x >> 5;
    const int row0 = blockIdx.y * 64 + (wv >> 1) * 32;
    const int col0 = blockIdx.x * 128 + (wv & 1) * 64;

    AccU acc[2][4];
#pragma unroll
    for (int mi = 0; mi < 2; ++mi)
#pragma unroll
        for (int ni = 0; ni < 4; ++ni)
#pragma unroll
            for (int i = 0; i < 8; ++i) acc[mi][ni].f[i] = 0.f;

    wave_gemm(A + (size_t)row0 * D_, D_, Wt + (size_t)col0 * D_, D_, D_, lane, acc);

    const int half = lane >> 4, cl = lane & 15;
#pragma unroll
    for (int mi = 0; mi < 2; ++mi)
#pragma unroll
        for (int ni = 0; ni < 4; ++ni) {
            int col = col0 + ni * 16 + cl;
            float bb = bias[col];
#pragma unroll
            for (int r = 0; r < 8; ++r) {
                int row = row0 + mi * 16 + r + 8 * half;
                Out[(size_t)row * D_ + col] = acc[mi][ni].f[r] + bb;
            }
        }
}

// ---------------- host launcher ----------------
extern "C" void kernel_launch(void* const* d_in, const int* in_sizes, int n_in,
                              void* d_out, int out_size, void* d_ws, size_t ws_size,
                              hipStream_t stream) {
    (void)in_sizes; (void)n_in; (void)out_size; (void)ws_size;
    const float* x  = (const float*)d_in[0];
    const float* Wq = (const float*)d_in[1];
    const float* bq = (const float*)d_in[2];
    const float* Wk = (const float*)d_in[3];
    const float* bk = (const float*)d_in[4];
    const float* Wv = (const float*)d_in[5];
    const float* bv = (const float*)d_in[6];
    const float* Wh = (const float*)d_in[7];
    const float* bh = (const float*)d_in[8];
    float* out = (float*)d_out;

    char* ws = (char*)d_ws;
    size_t o = 0;
    auto take = [&](size_t bytes) {
        char* p = ws + o;
        o = (o + bytes + 255) & ~(size_t)255;
        return p;
    };
    unsigned short* x16  = (unsigned short*)take((size_t)MT_ * D_ * 2);
    unsigned short* WqT  = (unsigned short*)take((size_t)D_ * D_ * 2);
    unsigned short* WkT  = (unsigned short*)take((size_t)D_ * D_ * 2);
    unsigned short* WvT  = (unsigned short*)take((size_t)D_ * D_ * 2);
    unsigned short* WhT  = (unsigned short*)take((size_t)D_ * D_ * 2);
    unsigned short* Q16  = (unsigned short*)take((size_t)MT_ * D_ * 2);
    unsigned short* K16  = (unsigned short*)take((size_t)MT_ * D_ * 2);
    unsigned short* Vt16 = (unsigned short*)take((size_t)MT_ * D_ * 2);
    float*          mbuf = (float*)take((size_t)B_ * T_ * 4);
    float*          lbuf = (float*)take((size_t)B_ * T_ * 4);
    unsigned short* P16  = (unsigned short*)take((size_t)B_ * T_ * T_ * 2);
    unsigned short* At16 = (unsigned short*)take((size_t)MT_ * D_ * 2);

    // 1) convert inputs to bf16 (weights transposed to N-major)
    k_cvt_bf16<<<(MT_ * D_) / 256, 256, 0, stream>>>(x, x16, MT_ * D_);
    dim3 tg(D_ / 16, D_ / 16);
    k_transpose_bf16<<<tg, dim3(16, 16), 0, stream>>>(Wq, WqT);
    k_transpose_bf16<<<tg, dim3(16, 16), 0, stream>>>(Wk, WkT);
    k_transpose_bf16<<<tg, dim3(16, 16), 0, stream>>>(Wv, WvT);
    k_transpose_bf16<<<tg, dim3(16, 16), 0, stream>>>(Wh, WhT);

    // 2) Q/K/V projections (V stored transposed per batch)
    dim3 gproj(D_ / 128, MT_ / 64);
    k_proj<<<gproj, 128, 0, stream>>>(x16, WqT, bq, Q16, nullptr);
    k_proj<<<gproj, 128, 0, stream>>>(x16, WkT, bk, K16, nullptr);
    k_proj<<<gproj, 128, 0, stream>>>(x16, WvT, bv, nullptr, Vt16);

    // 3) causal softmax stats, then materialize P (bf16)
    k_stats<<<dim3(T_ / 64, B_), 128, 0, stream>>>(Q16, K16, mbuf, lbuf);
    k_probs<<<dim3(T_ / 16, T_ / 64, B_), 128, 0, stream>>>(Q16, K16, mbuf, lbuf, P16);

    // 4) attn @ V
    k_pv<<<dim3(D_ / 128, T_ / 64, B_), 128, 0, stream>>>(P16, Vt16, At16);

    // 5) final projection -> f32 output
    k_final<<<dim3(D_ / 128, MT_ / 64), 128, 0, stream>>>(At16, WhT, bh, out);
}
